// PointNet2Feature_30356828848547
// MI455X (gfx1250) — matrix operand order repaired and checked by reading, not compiled, not run on hardware
//
#include <hip/hip_runtime.h>
#include <hip/hip_bf16.h>
#include <math.h>

// ---------------------------------------------------------------------------
// PointNet++ feature net for MI455X (gfx1250).
// All einsums -> V_WMMA_F32_16X16X32_F16 over [col][Kpad]-major f16 activations.
// Round-3:
//  * GEMM BN mode is a template parameter -> branch-free inner loop, the
//    scheduler can unroll + software-pipeline loads against wmma again.
//  * Fused BN done as packed f16 FMA (b = b*scale + shift, scale/shift
//    precomputed by the stats kernel) -> v_pk_fma_f16/v_pk_max instead of
//    f32 cvt chains; co-executes with XDL wmma.
//  * 4 M-tiles per wave register blocking; BN+ReLU fused into consumers.
// Workspace use ~315 MB.
// ---------------------------------------------------------------------------

typedef _Float16 h16;
typedef __attribute__((ext_vector_type(16))) _Float16 v16h;
typedef __attribute__((ext_vector_type(8)))  _Float16 v8h;
typedef __attribute__((ext_vector_type(8)))  float    v8f;

#define BB   16
#define NN   4096
#define EPSF 1e-5f

// ======================= WMMA GEMM (4 M-tiles / wave) ======================
// H[col][O] = W[O][Kpad] x BN?(X[col][Kpad])
// Requires: O % 64 == 0, Kpad % 32 == 0, cols % 16 == 0.
// MODE: 0 = raw, 2 = fused BN+ReLU on B operand (b = max(b*sc+sh, 0)).
// Fragment layouts per CDNA5 ISA 7.12.2:
//   A lane l: m=l&15, kbase=(l>>4)*8 ; v[0..3]=K kbase..+7, v[4..7]=+16
//   B lane l: n=l&15, kbase=(l>>4)*8 ; v[0..3]=K kbase..+7, v[4..7]=+16
//   D lane l: n=l&15, vgpr r -> M = (l>>4)*8 + r
template <int MODE>
__global__ __launch_bounds__(256)
void k_wmma_gemm(const h16* __restrict__ W, const h16* __restrict__ X,
                 h16* __restrict__ H, int O, int Kpad, int cols,
                 const h16* __restrict__ bnsc, const h16* __restrict__ bnsh) {
  int gid  = blockIdx.x * blockDim.x + threadIdx.x;
  int wave = gid >> 5;
  int lane = gid & 31;
  int gM   = O >> 6;          // groups of 4 M-tiles
  int tN   = cols >> 4;
  if (wave >= gM * tN) return;
  int tm4  = wave % gM;       // which group of 64 output channels
  int tn   = wave / gM;       // which column tile
  int half = lane >> 4;
  int l16  = lane & 15;
  const h16* wp = W + (size_t)((tm4 << 6) + l16) * Kpad + half * 8;
  const h16* xp = X + (size_t)((tn  << 4) + l16) * Kpad + half * 8;
  union F16x16 { v16h v; v8h h[2]; };
  v8f acc[4] = {};
  for (int k = 0; k < Kpad; k += 32) {
    __builtin_prefetch(xp + k + 32, 0, 1);
    F16x16 b;
    b.h[0] = *(const v8h*)(xp + k);
    b.h[1] = *(const v8h*)(xp + k + 16);
    if (MODE != 0) {                   // fused BN(+ReLU), packed f16 FMA
      int kb = k + half * 8;
      v8h s0 = *(const v8h*)(bnsc + kb);
      v8h s1 = *(const v8h*)(bnsc + kb + 16);
      v8h t0 = *(const v8h*)(bnsh + kb);
      v8h t1 = *(const v8h*)(bnsh + kb + 16);
      b.h[0] = b.h[0] * s0 + t0;
      b.h[1] = b.h[1] * s1 + t1;
      if (MODE == 2) {
        v8h z = {};
        b.h[0] = __builtin_elementwise_max(b.h[0], z);
        b.h[1] = __builtin_elementwise_max(b.h[1], z);
      }
    }
    F16x16 a0, a1, a2, a3;
    const h16* w0 = wp + k;
    a0.h[0] = *(const v8h*)(w0);
    a0.h[1] = *(const v8h*)(w0 + 16);
    a1.h[0] = *(const v8h*)(w0 + (size_t)16 * Kpad);
    a1.h[1] = *(const v8h*)(w0 + (size_t)16 * Kpad + 16);
    a2.h[0] = *(const v8h*)(w0 + (size_t)32 * Kpad);
    a2.h[1] = *(const v8h*)(w0 + (size_t)32 * Kpad + 16);
    a3.h[0] = *(const v8h*)(w0 + (size_t)48 * Kpad);
    a3.h[1] = *(const v8h*)(w0 + (size_t)48 * Kpad + 16);
    acc[0] = __builtin_amdgcn_wmma_f32_16x16x32_f16(false, a0.v, false, b.v,
                                                    (short)0, acc[0], false, false);
    acc[1] = __builtin_amdgcn_wmma_f32_16x16x32_f16(false, a1.v, false, b.v,
                                                    (short)0, acc[1], false, false);
    acc[2] = __builtin_amdgcn_wmma_f32_16x16x32_f16(false, a2.v, false, b.v,
                                                    (short)0, acc[2], false, false);
    acc[3] = __builtin_amdgcn_wmma_f32_16x16x32_f16(false, a3.v, false, b.v,
                                                    (short)0, acc[3], false, false);
  }
  h16* hp = H + (size_t)((tn << 4) + l16) * O + (tm4 << 6) + half * 8;
#pragma unroll
  for (int i = 0; i < 4; ++i) {
    v8h o;
#pragma unroll
    for (int r = 0; r < 8; ++r) o[r] = (h16)acc[i][r];
    *(v8h*)(hp + i * 16) = o;
  }
}

// ======================= BN stats (per-channel over all cols) ==============
// 16 channels/block; lane%16 = channel -> contiguous 32B reads per 16-lane
// group. Emits f32 mean/rstd (for maxpool) and f16 scale/shift (for GEMM):
//   scale = rstd, shift = -mean*rstd  so  bn(x) = x*scale + shift.
__global__ void k_bn_stats(const h16* __restrict__ H, float* __restrict__ mean,
                           float* __restrict__ rstd, h16* __restrict__ sc,
                           h16* __restrict__ sh, int O, int cols) {
  int tid = threadIdx.x;
  int c = (blockIdx.x << 4) + (tid & 15);
  float s = 0.f, q = 0.f;
  for (int col = tid >> 4; col < cols; col += 16) {
    float v = (float)H[(size_t)col * O + c];
    s += v; q += v * v;
  }
  __shared__ float ls[256], lq[256];
  ls[tid] = s; lq[tid] = q;
  __syncthreads();
  for (int off = 128; off >= 16; off >>= 1) {
    if (tid < off) { ls[tid] += ls[tid + off]; lq[tid] += lq[tid + off]; }
    __syncthreads();
  }
  if (tid < 16) {
    float m = ls[tid] / (float)cols;
    float var = lq[tid] / (float)cols - m * m;
    float r = rsqrtf(var + EPSF);
    mean[c] = m;
    rstd[c] = r;
    sc[c] = (h16)r;
    sh[c] = (h16)(-m * r);
  }
}

// ======== max over groups of S consecutive cols, fused BN epilogue =========
// BN and ReLU are monotone nondecreasing (rstd>0), so f(max) == max(f).
__global__ void k_maxpool(const h16* __restrict__ H, float* __restrict__ out,
                          int O, int S, int G,
                          const float* __restrict__ mean,
                          const float* __restrict__ rstd, int relu) {
  int i = blockIdx.x * blockDim.x + threadIdx.x;
  if (i >= G * O) return;
  int g = i / O, c = i % O;
  const h16* p = H + (size_t)g * S * O + c;
  float m = -3.4e38f;
  for (int s = 0; s < S; ++s) m = fmaxf(m, (float)p[(size_t)s * O]);
  float v = (m - mean[c]) * rstd[c];
  if (relu) v = fmaxf(v, 0.f);
  out[i] = v;
}

// ======================= weight pad/convert f32 -> f16 =====================
__global__ void k_cvt_w(const float* __restrict__ w, h16* __restrict__ wf,
                        int O, int K, int Kpad) {
  int i = blockIdx.x * blockDim.x + threadIdx.x;
  if (i >= O * Kpad) return;
  int o = i / Kpad, k = i % Kpad;
  wf[i] = (k < K) ? (h16)w[o * K + k] : (h16)0.f;
}

// ======================= qnet input packing (K: 3 -> 32) ===================
__global__ void k_pack_qnet(const float* __restrict__ xa, h16* __restrict__ X0) {
  int i = blockIdx.x * blockDim.x + threadIdx.x;  // over B*N*32
  if (i >= BB * NN * 32) return;
  int k = i & 31;
  int col = i >> 5;                 // b*N + n
  int b = col >> 12, n = col & (NN - 1);
  X0[i] = (k < 3) ? (h16)xa[(size_t)(b * 3 + k) * NN + n] : (h16)0.f;
}

// ======================= small dense layers ================================
__global__ void k_fc(const float* __restrict__ X, const float* __restrict__ W,
                     const float* __restrict__ bias, float* __restrict__ Y,
                     int Bn, int O, int K) {
  int i = blockIdx.x * blockDim.x + threadIdx.x;
  if (i >= Bn * O) return;
  int b = i / O, o = i % O;
  const float* x = X + (size_t)b * K;
  const float* w = W + (size_t)o * K;
  float s = bias ? bias[o] : 0.f;
  for (int k = 0; k < K; ++k) s += x[k] * w[k];
  Y[i] = s;
}

__global__ void k_bn_rows(float* __restrict__ Y, int Bn, int O, int relu) {
  int o = blockIdx.x * blockDim.x + threadIdx.x;
  if (o >= O) return;
  float s = 0.f, s2 = 0.f;
  for (int b = 0; b < Bn; ++b) { float v = Y[b * O + o]; s += v; s2 += v * v; }
  float m = s / Bn, var = s2 / Bn - m * m, r = rsqrtf(var + EPSF);
  for (int b = 0; b < Bn; ++b) {
    float v = (Y[b * O + o] - m) * r;
    Y[b * O + o] = relu ? fmaxf(v, 0.f) : v;
  }
}

// ======================= quaternion -> rotation, rotate cloud ==============
__global__ void k_quat_rot(const float* __restrict__ q4, float* __restrict__ R) {
  int b = blockIdx.x * blockDim.x + threadIdx.x;
  if (b >= BB) return;
  float w = q4[b * 4 + 0] + 1.f, x = q4[b * 4 + 1];
  float y = q4[b * 4 + 2],       z = q4[b * 4 + 3];
  float inv = rsqrtf(w * w + x * x + y * y + z * z);
  w *= inv; x *= inv; y *= inv; z *= inv;
  float* r = R + b * 9;
  r[0] = 1.f - 2.f * (y * y + z * z); r[1] = 2.f * (x * y - w * z); r[2] = 2.f * (x * z + w * y);
  r[3] = 2.f * (x * y + w * z); r[4] = 1.f - 2.f * (x * x + z * z); r[5] = 2.f * (y * z - w * x);
  r[6] = 2.f * (x * z - w * y); r[7] = 2.f * (y * z + w * x); r[8] = 1.f - 2.f * (x * x + y * y);
}

__global__ void k_rotate(const float* __restrict__ xa, const float* __restrict__ R,
                         float* __restrict__ xyz) {  // xyz[b][n][3]
  int i = blockIdx.x * blockDim.x + threadIdx.x;     // over B*N
  if (i >= BB * NN) return;
  int b = i >> 12, n = i & (NN - 1);
  float p0 = xa[(size_t)(b * 3 + 0) * NN + n];
  float p1 = xa[(size_t)(b * 3 + 1) * NN + n];
  float p2 = xa[(size_t)(b * 3 + 2) * NN + n];
  const float* r = R + b * 9;
  xyz[(size_t)i * 3 + 0] = r[0] * p0 + r[1] * p1 + r[2] * p2;
  xyz[(size_t)i * 3 + 1] = r[3] * p0 + r[4] * p1 + r[5] * p2;
  xyz[(size_t)i * 3 + 2] = r[6] * p0 + r[7] * p1 + r[8] * p2;
}

// ======================= farthest point sampling ===========================
// One block per batch. Emits current `far` then updates (matches lax.scan).
__global__ void k_fps(const float* __restrict__ xyz, int n, int npoint,
                      int* __restrict__ out) {
  int b = blockIdx.x;
  int tid = threadIdx.x;
  __shared__ float dist[4096];
  __shared__ float rv[256];
  __shared__ int   ri[256];
  __shared__ int   sfar;
  for (int i = tid; i < n; i += blockDim.x) dist[i] = 1e10f;
  if (tid == 0) sfar = 0;
  __syncthreads();
  const float* base = xyz + (size_t)b * n * 3;
  for (int it = 0; it < npoint; ++it) {
    int far = sfar;
    if (tid == 0) out[b * npoint + it] = far;
    float cx = base[far * 3], cy = base[far * 3 + 1], cz = base[far * 3 + 2];
    float bm = -1.f; int ba = 0x7fffffff;
    for (int i = tid; i < n; i += blockDim.x) {
      float dx = base[i * 3] - cx, dy = base[i * 3 + 1] - cy, dz = base[i * 3 + 2] - cz;
      float d = dx * dx + dy * dy + dz * dz;
      float nd = fminf(dist[i], d);
      dist[i] = nd;
      if (nd > bm || (nd == bm && i < ba)) { bm = nd; ba = i; }
    }
    rv[tid] = bm; ri[tid] = ba;
    __syncthreads();
    for (int off = 128; off > 0; off >>= 1) {
      if (tid < off) {
        if (rv[tid + off] > rv[tid] ||
            (rv[tid + off] == rv[tid] && ri[tid + off] < ri[tid])) {
          rv[tid] = rv[tid + off]; ri[tid] = ri[tid + off];
        }
      }
      __syncthreads();
    }
    if (tid == 0) sfar = ri[0];
    __syncthreads();
  }
}

// ======================= ball query + grouping =============================
// One thread per centroid. Ascending scan == sorted candidates; pad w/ first.
__global__ void k_ball_group(const float* __restrict__ xyz, int n,
                             const int* __restrict__ fpsidx, int np,
                             const float* __restrict__ feat, int C,
                             float r2, int ns,
                             h16* __restrict__ Xg, int Kpad,
                             float* __restrict__ newxyz) {
  int i = blockIdx.x * blockDim.x + threadIdx.x;  // over B*np
  if (i >= BB * np) return;
  int b = i / np;
  const float* base = xyz + (size_t)b * n * 3;
  int fi = fpsidx[i];
  float cx = base[fi * 3], cy = base[fi * 3 + 1], cz = base[fi * 3 + 2];
  newxyz[(size_t)i * 3 + 0] = cx;
  newxyz[(size_t)i * 3 + 1] = cy;
  newxyz[(size_t)i * 3 + 2] = cz;
  int idx[32];
  int cnt = 0;
  for (int j = 0; j < n && cnt < ns; ++j) {
    float dx = base[j * 3] - cx, dy = base[j * 3 + 1] - cy, dz = base[j * 3 + 2] - cz;
    if (dx * dx + dy * dy + dz * dz < r2) idx[cnt++] = j;
  }
  if (cnt == 0) { idx[0] = 0; cnt = 1; }
  for (int s = cnt; s < ns; ++s) idx[s] = idx[0];
  h16* xg = Xg + (size_t)i * ns * Kpad;
  for (int s = 0; s < ns; ++s) {
    int j = idx[s];
    h16* col = xg + (size_t)s * Kpad;
    col[0] = (h16)(base[j * 3] - cx);
    col[1] = (h16)(base[j * 3 + 1] - cy);
    col[2] = (h16)(base[j * 3 + 2] - cz);
    int k = 3;
    if (feat) {
      const float* f = feat + (size_t)(b * n + j) * C;
      for (int c = 0; c < C; ++c) col[k++] = (h16)f[c];
    }
    for (; k < Kpad; ++k) col[k] = (h16)0.f;
  }
}

// ======================= SA3 group-all (K: 259 -> 288) =====================
__global__ void k_group_all(const float* __restrict__ xyz2,
                            const float* __restrict__ f2,
                            h16* __restrict__ Xg) {
  int i = blockIdx.x * blockDim.x + threadIdx.x;  // over B*64
  if (i >= BB * 64) return;
  h16* col = Xg + (size_t)i * 288;
  col[0] = (h16)xyz2[(size_t)i * 3 + 0];
  col[1] = (h16)xyz2[(size_t)i * 3 + 1];
  col[2] = (h16)xyz2[(size_t)i * 3 + 2];
  const float* f = f2 + (size_t)i * 256;
  for (int c = 0; c < 256; ++c) col[3 + c] = (h16)f[c];
  for (int k = 259; k < 288; ++k) col[k] = (h16)0.f;
}

// ======================= output L2 normalize ===============================
__global__ void k_norm_out(const float* __restrict__ Y, float* __restrict__ out,
                           int Bn, int D) {
  int b = blockIdx.x * blockDim.x + threadIdx.x;
  if (b >= Bn) return;
  float s = 0.f;
  for (int d = 0; d < D; ++d) { float v = Y[b * D + d]; s += v * v; }
  float inv = rsqrtf(s);
  for (int d = 0; d < D; ++d) out[b * D + d] = Y[b * D + d] * inv;
}

// ===========================================================================
static inline int cdiv(long a, long b) { return (int)((a + b - 1) / b); }

static void gemm(const h16* W, const h16* X, h16* H, int O, int Kpad, int cols,
                 const h16* sc, const h16* sh, int mode, hipStream_t s) {
  long thr = (long)(O / 64) * (cols / 16) * 32;
  int g = cdiv(thr, 256);
  if (mode == 0)
    k_wmma_gemm<0><<<g, 256, 0, s>>>(W, X, H, O, Kpad, cols, sc, sh);
  else
    k_wmma_gemm<2><<<g, 256, 0, s>>>(W, X, H, O, Kpad, cols, sc, sh);
}

extern "C" void kernel_launch(void* const* d_in, const int* in_sizes, int n_in,
                              void* d_out, int out_size, void* d_ws, size_t ws_size,
                              hipStream_t stream) {
  const float* xa    = (const float*)d_in[0];
  const float* qc1   = (const float*)d_in[1];
  const float* qc2   = (const float*)d_in[2];
  const float* qc3   = (const float*)d_in[3];
  const float* qf1   = (const float*)d_in[4];
  const float* qf2   = (const float*)d_in[5];
  const float* qf3w  = (const float*)d_in[6];
  const float* qf3b  = (const float*)d_in[7];
  const float* s1w1  = (const float*)d_in[8];
  const float* s1w2  = (const float*)d_in[9];
  const float* s1w3  = (const float*)d_in[10];
  const float* s2w1  = (const float*)d_in[11];
  const float* s2w2  = (const float*)d_in[12];
  const float* s2w3  = (const float*)d_in[13];
  const float* s3w1  = (const float*)d_in[14];
  const float* s3w2  = (const float*)d_in[15];
  const float* s3w3  = (const float*)d_in[16];
  const float* fc1   = (const float*)d_in[17];
  const float* fc2   = (const float*)d_in[18];
  const float* fc3w  = (const float*)d_in[19];
  const float* fc3b  = (const float*)d_in[20];
  (void)in_sizes; (void)n_in; (void)out_size; (void)ws_size;

  char* base = (char*)d_ws;
  size_t off = 0;
  auto alloc = [&](size_t bytes) -> char* {
    off = (off + 255) & ~(size_t)255;
    char* p = base + off;
    off += bytes;
    return p;
  };

  const int COLS = BB * NN;  // 65536
  // ---- f16 weight buffers (O x Kpad)
  h16* Wq1  = (h16*)alloc((size_t)256 * 32 * 2);
  h16* Wq2  = (h16*)alloc((size_t)512 * 256 * 2);
  h16* Wq3  = (h16*)alloc((size_t)1024 * 512 * 2);
  h16* W11  = (h16*)alloc((size_t)128 * 32 * 2);
  h16* W12  = (h16*)alloc((size_t)128 * 128 * 2);
  h16* W13  = (h16*)alloc((size_t)128 * 128 * 2);
  h16* W21  = (h16*)alloc((size_t)256 * 160 * 2);
  h16* W22  = (h16*)alloc((size_t)256 * 256 * 2);
  h16* W23  = (h16*)alloc((size_t)256 * 256 * 2);
  h16* W31  = (h16*)alloc((size_t)512 * 288 * 2);
  h16* W32  = (h16*)alloc((size_t)512 * 512 * 2);
  h16* W33  = (h16*)alloc((size_t)1024 * 512 * 2);
  // ---- qnet activations (raw, pre-BN; BN fused into consumers)
  h16* X0   = (h16*)alloc((size_t)COLS * 32 * 2);
  h16* H1   = (h16*)alloc((size_t)COLS * 256 * 2);
  h16* H2   = (h16*)alloc((size_t)COLS * 512 * 2);
  h16* H3   = (h16*)alloc((size_t)COLS * 1024 * 2);
  float* mean = (float*)alloc(1024 * 4);
  float* rstd = (float*)alloc(1024 * 4);
  h16*  bnsc = (h16*)alloc(1024 * 2);
  h16*  bnsh = (h16*)alloc(1024 * 2);
  float* gmax = (float*)alloc((size_t)BB * 1024 * 4);
  float* t512 = (float*)alloc((size_t)BB * 512 * 4);
  float* t256 = (float*)alloc((size_t)BB * 256 * 4);
  float* q4   = (float*)alloc((size_t)BB * 4 * 4);
  float* Rm   = (float*)alloc((size_t)BB * 9 * 4);
  float* xyz  = (float*)alloc((size_t)BB * NN * 3 * 4);
  // ---- SA1
  int*   fps1  = (int*)alloc((size_t)BB * 128 * 4);
  float* nxyz1 = (float*)alloc((size_t)BB * 128 * 3 * 4);
  h16*   Xg1   = (h16*)alloc((size_t)COLS * 32 * 2);
  h16*   HA    = (h16*)alloc((size_t)COLS * 128 * 2);
  h16*   HB    = (h16*)alloc((size_t)COLS * 128 * 2);
  float* f1    = (float*)alloc((size_t)BB * 128 * 128 * 4);
  // ---- SA2
  int*   fps2  = (int*)alloc((size_t)BB * 64 * 4);
  float* nxyz2 = (float*)alloc((size_t)BB * 64 * 3 * 4);
  h16*   Xg2   = (h16*)alloc((size_t)16384 * 160 * 2);
  h16*   HA2   = (h16*)alloc((size_t)16384 * 256 * 2);
  h16*   HB2   = (h16*)alloc((size_t)16384 * 256 * 2);
  float* f2    = (float*)alloc((size_t)BB * 64 * 256 * 4);
  // ---- SA3 (group all)
  h16*   Xg3   = (h16*)alloc((size_t)1024 * 288 * 2);
  h16*   G1    = (h16*)alloc((size_t)1024 * 512 * 2);
  h16*   G2    = (h16*)alloc((size_t)1024 * 512 * 2);
  h16*   G3    = (h16*)alloc((size_t)1024 * 1024 * 2);
  float* f3    = (float*)alloc((size_t)BB * 1024 * 4);
  // ---- head
  float* Y1 = (float*)alloc((size_t)BB * 512 * 4);
  float* Y2 = (float*)alloc((size_t)BB * 256 * 4);
  float* Y3 = (float*)alloc((size_t)BB * 32 * 4);

  auto stats = [&](const h16* H, int O, int cols) {
    k_bn_stats<<<O / 16, 256, 0, stream>>>(H, mean, rstd, bnsc, bnsh, O, cols);
  };

  // ================= weight conversion =================
  auto cvt = [&](const float* w, h16* wf, int O, int K, int Kpad) {
    k_cvt_w<<<cdiv((long)O * Kpad, 256), 256, 0, stream>>>(w, wf, O, K, Kpad);
  };
  cvt(qc1, Wq1, 256, 3, 32);     cvt(qc2, Wq2, 512, 256, 256);
  cvt(qc3, Wq3, 1024, 512, 512);
  cvt(s1w1, W11, 128, 3, 32);    cvt(s1w2, W12, 128, 128, 128);
  cvt(s1w3, W13, 128, 128, 128);
  cvt(s2w1, W21, 256, 131, 160); cvt(s2w2, W22, 256, 256, 256);
  cvt(s2w3, W23, 256, 256, 256);
  cvt(s3w1, W31, 512, 259, 288); cvt(s3w2, W32, 512, 512, 512);
  cvt(s3w3, W33, 1024, 512, 512);

  // ================= qnet =================
  k_pack_qnet<<<cdiv((long)COLS * 32, 256), 256, 0, stream>>>(xa, X0);
  gemm(Wq1, X0, H1, 256, 32, COLS, nullptr, nullptr, 0, stream);
  stats(H1, 256, COLS);
  gemm(Wq2, H1, H2, 512, 256, COLS, bnsc, bnsh, 2, stream);
  stats(H2, 512, COLS);
  gemm(Wq3, H2, H3, 1024, 512, COLS, bnsc, bnsh, 2, stream);
  stats(H3, 1024, COLS);
  k_maxpool<<<cdiv((long)BB * 1024, 256), 256, 0, stream>>>(
      H3, gmax, 1024, NN, BB, mean, rstd, 0);

  k_fc<<<cdiv((long)BB * 512, 128), 128, 0, stream>>>(gmax, qf1, nullptr, t512, BB, 512, 1024);
  k_bn_rows<<<cdiv(512, 128), 128, 0, stream>>>(t512, BB, 512, 1);
  k_fc<<<cdiv((long)BB * 256, 128), 128, 0, stream>>>(t512, qf2, nullptr, t256, BB, 256, 512);
  k_bn_rows<<<cdiv(256, 128), 128, 0, stream>>>(t256, BB, 256, 1);
  k_fc<<<1, 64, 0, stream>>>(t256, qf3w, qf3b, q4, BB, 4, 256);
  k_quat_rot<<<1, 32, 0, stream>>>(q4, Rm);
  k_rotate<<<cdiv((long)BB * NN, 256), 256, 0, stream>>>(xa, Rm, xyz);

  // ================= SA1: npoint=128, r=0.2, ns=32 =================
  k_fps<<<BB, 256, 0, stream>>>(xyz, NN, 128, fps1);
  k_ball_group<<<cdiv((long)BB * 128, 128), 128, 0, stream>>>(
      xyz, NN, fps1, 128, nullptr, 0, 0.04f, 32, Xg1, 32, nxyz1);
  gemm(W11, Xg1, HA, 128, 32, COLS, nullptr, nullptr, 0, stream);
  stats(HA, 128, COLS);
  gemm(W12, HA, HB, 128, 128, COLS, bnsc, bnsh, 2, stream);
  stats(HB, 128, COLS);
  gemm(W13, HB, HA, 128, 128, COLS, bnsc, bnsh, 2, stream);
  stats(HA, 128, COLS);
  k_maxpool<<<cdiv((long)BB * 128 * 128, 256), 256, 0, stream>>>(
      HA, f1, 128, 32, BB * 128, mean, rstd, 1);

  // ================= SA2: npoint=64, r=0.4, ns=16, C=128 =================
  k_fps<<<BB, 256, 0, stream>>>(nxyz1, 128, 64, fps2);
  k_ball_group<<<cdiv((long)BB * 64, 128), 128, 0, stream>>>(
      nxyz1, 128, fps2, 64, f1, 128, 0.16f, 16, Xg2, 160, nxyz2);
  gemm(W21, Xg2, HA2, 256, 160, 16384, nullptr, nullptr, 0, stream);
  stats(HA2, 256, 16384);
  gemm(W22, HA2, HB2, 256, 256, 16384, bnsc, bnsh, 2, stream);
  stats(HB2, 256, 16384);
  gemm(W23, HB2, HA2, 256, 256, 16384, bnsc, bnsh, 2, stream);
  stats(HA2, 256, 16384);
  k_maxpool<<<cdiv((long)BB * 64 * 256, 256), 256, 0, stream>>>(
      HA2, f2, 256, 16, BB * 64, mean, rstd, 1);

  // ================= SA3: group all over 64 points =================
  k_group_all<<<cdiv((long)BB * 64, 128), 128, 0, stream>>>(nxyz2, f2, Xg3);
  gemm(W31, Xg3, G1, 512, 288, 1024, nullptr, nullptr, 0, stream);
  stats(G1, 512, 1024);
  gemm(W32, G1, G2, 512, 512, 1024, bnsc, bnsh, 2, stream);
  stats(G2, 512, 1024);
  gemm(W33, G2, G3, 1024, 512, 1024, bnsc, bnsh, 2, stream);
  stats(G3, 1024, 1024);
  k_maxpool<<<cdiv((long)BB * 1024, 256), 256, 0, stream>>>(
      G3, f3, 1024, 64, BB, mean, rstd, 1);

  // ================= head =================
  k_fc<<<cdiv((long)BB * 512, 128), 128, 0, stream>>>(f3, fc1, nullptr, Y1, BB, 512, 1024);
  k_bn_rows<<<cdiv(512, 128), 128, 0, stream>>>(Y1, BB, 512, 1);
  k_fc<<<cdiv((long)BB * 256, 128), 128, 0, stream>>>(Y1, fc2, nullptr, Y2, BB, 256, 512);
  k_bn_rows<<<cdiv(256, 128), 128, 0, stream>>>(Y2, BB, 256, 1);
  k_fc<<<cdiv((long)BB * 32, 128), 128, 0, stream>>>(Y2, fc3w, fc3b, Y3, BB, 32, 256);
  k_norm_out<<<1, 32, 0, stream>>>(Y3, (float*)d_out, BB, 32);
}
